// Encoder_21715354649978
// MI455X (gfx1250) — matrix operand order repaired and checked by reading, not compiled, not run on hardware
//
#include <hip/hip_runtime.h>
#include <hip/hip_bf16.h>

// ---------------------------------------------------------------------------
// THP encoder layer for MI455X (gfx1250): bf16 WMMA GEMMs + flash attention,
// async global->LDS staging (ASYNCcnt) with double buffering.
// B=4 S=2048 DM=512 H=4 DK=DV=128 DI=1024; R = B*S = 8192 rows.
// ---------------------------------------------------------------------------

typedef __attribute__((ext_vector_type(16))) __bf16 v16bf;
typedef __attribute__((ext_vector_type(8)))  float  v8f;
typedef __attribute__((ext_vector_type(4)))  int    v4i;

#define S_LEN   2048
#define DMODEL  512
#define NHEAD   4
#define DHEAD   128
#define NVTX    5000
#define ROWS    8192        // B * S

// CDNA5 async global->LDS copies (ASYNCcnt). Guarded: fall back to
// register-staged copies if the toolchain lacks the builtins.
#if __has_builtin(__builtin_amdgcn_global_load_async_to_lds_b128) && \
    __has_builtin(__builtin_amdgcn_s_wait_asynccnt)
#define USE_ASYNC 1
#else
#define USE_ASYNC 0
#endif

#if USE_ASYNC
// builtin signature (probe-confirmed by diagnostic): (v4i global*, v4i lds*, imm, imm)
typedef __attribute__((address_space(1))) v4i* gv4p;
typedef __attribute__((address_space(3))) v4i* lv4p;
__device__ __forceinline__ void async_ld128(const void* g, void* l) {
  __builtin_amdgcn_global_load_async_to_lds_b128((gv4p)g, (lv4p)l, 0, 0);
}
__device__ __forceinline__ void async_wait0() {
  __builtin_amdgcn_s_wait_asynccnt(0);
}
#endif

__device__ __forceinline__ v8f wmma_bf16(v16bf a, v16bf b, v8f c) {
  // D = A(16x32) * B(32x16) + C(16x16), f32 accumulate
  return __builtin_amdgcn_wmma_f32_16x16x32_bf16(false, a, false, b,
                                                 (short)0, c, false, false);
}

__device__ __forceinline__ v16bf pack_bf16(uint4 a, uint4 b) {
  union { uint4 u[2]; v16bf v; } t;
  t.u[0] = a; t.u[1] = b;
  return t.v;
}

// ---------------------------------------------------------------------------
// f32 [K,N] row-major  ->  bf16 [N,K] (transposed), 32x32 LDS tiles so both
// the global read and the global write are coalesced.
// ---------------------------------------------------------------------------
__global__ __launch_bounds__(256)
void cvtT_kernel(const float* __restrict__ src, __bf16* __restrict__ dst,
                 int K, int N) {
  __shared__ float t[32][33];
  int k0 = blockIdx.x * 32, n0 = blockIdx.y * 32;
  int tx = threadIdx.x & 31, ty = threadIdx.x >> 5;   // ty in 0..7
  for (int i = ty; i < 32; i += 8)
    t[i][tx] = src[(size_t)(k0 + i) * N + n0 + tx];
  __syncthreads();
  for (int i = ty; i < 32; i += 8)
    dst[(size_t)(n0 + i) * K + k0 + tx] = (__bf16)t[tx][i];
}

// ---------------------------------------------------------------------------
// enc0 = event_emb[type] + vertex_emb[vertex] + temporal_enc   (f32 + bf16)
// one block per row; each thread does an even/odd (sin,cos) pair sharing pv.
// ---------------------------------------------------------------------------
__global__ __launch_bounds__(256)
void embed_kernel(const int* __restrict__ et, const int* __restrict__ vx,
                  const float* __restrict__ tim, const float* __restrict__ msk,
                  const float* __restrict__ eemb, const float* __restrict__ vemb,
                  float* __restrict__ encf, __bf16* __restrict__ encb) {
  int row = blockIdx.x;
  int e  = et[row];
  int vv = vx[row];
  float t  = tim[row];
  float mk = msk[row];
  int d = threadIdx.x * 2;                 // 256 threads * 2 = 512 columns
  float pv = __powf(10000.f, (float)d * (1.f / (float)DMODEL));
  float r  = t / pv;
  float2 ee = *(const float2*)(eemb + e * DMODEL + d);
  float2 ve = *(const float2*)(vemb + vv * DMODEL + d);
  float v0 = ee.x + ve.x + __sinf(r) * mk;
  float v1 = ee.y + ve.y + __cosf(r) * mk;
  size_t o = (size_t)row * DMODEL + d;
  *(float2*)(encf + o) = make_float2(v0, v1);
  union { __bf16 h[2]; unsigned u; } pk;
  pk.h[0] = (__bf16)v0; pk.h[1] = (__bf16)v1;
  *(unsigned*)(encb + o) = pk.u;
}

// ---------------------------------------------------------------------------
// sim[b,i,j] = (vi&&vj) ? A[vi-1,vj-1] * W[vi-1,vj-1] * 10 : 0   (x4 per thread)
// ---------------------------------------------------------------------------
__global__ __launch_bounds__(256)
void sim_kernel(const int* __restrict__ vx, const float* __restrict__ Am,
                const float* __restrict__ Wm, float* __restrict__ simo) {
  size_t idx4 = ((size_t)blockIdx.x * 256 + threadIdx.x) * 4;
  int j = (int)(idx4 & (S_LEN - 1));
  int i = (int)((idx4 >> 11) & (S_LEN - 1));
  int b = (int)(idx4 >> 22);
  int  vi  = vx[b * S_LEN + i];
  int4 vj4 = *(const int4*)(vx + b * S_LEN + j);
  float4 out;
  float* po = &out.x;
  const int* pj = &vj4.x;
  for (int q = 0; q < 4; ++q) {
    float v = 0.f;
    int vj = pj[q];
    if (vi != 0 && vj != 0) {
      size_t o = (size_t)(vi - 1) * NVTX + (size_t)(vj - 1);
      v = Am[o] * Wm[o] * 10.f;
    }
    po[q] = v;
  }
  *(float4*)(simo + idx4) = out;
}

// ---------------------------------------------------------------------------
// bf16 WMMA GEMM: C[M,N] = A[M,K] @ Bt[N,K]^T (+bias)(+res)(ReLU)
// Block 128x64, 8 waves, wave = 32x32 (2x2 WMMA tiles), BK=32.
// Double-buffered LDS; async global->LDS when available.
// OBT: write bf16 output transposed to [B, NHEAD, DHEAD, S] (d-major V).
// ---------------------------------------------------------------------------
template<bool BIAS, bool RELU, bool RES, bool OF32, bool OBF, bool OBT>
__global__ __launch_bounds__(256)
void gemm_bf16_kernel(const __bf16* __restrict__ Ag, const __bf16* __restrict__ Bt,
                      const float* __restrict__ bias, const float* __restrict__ res,
                      float* __restrict__ outf, __bf16* __restrict__ outb,
                      int M, int N, int K) {
  __shared__ __align__(16) __bf16 Al[2][128 * 32];   // [m][k]
  __shared__ __align__(16) __bf16 Bl[2][64 * 32];    // [n][k]

  const int tid  = threadIdx.x;
  const int wave = tid >> 5;
  const int lane = tid & 31;
  const int wm = wave >> 1;        // 0..3
  const int wn = wave & 1;         // 0..1
  const int hi = (lane >> 4) * 8;  // 0 or 8
  const int m0 = blockIdx.y * 128;
  const int n0 = blockIdx.x * 64;

  const int arow = tid >> 1, akh = (tid & 1) * 16;   // A: 16 bf16 per thread
  const int bcol = tid >> 2, bkg = (tid & 3) * 8;    // B: 8 bf16 per thread
  const __bf16* asrc = Ag + (size_t)(m0 + arow) * K + akh;
  const __bf16* bsrc = Bt + (size_t)(n0 + bcol) * K + bkg;

  v8f acc[2][2];
  for (int a = 0; a < 2; ++a)
    for (int bb = 0; bb < 2; ++bb)
      for (int r = 0; r < 8; ++r) acc[a][bb][r] = 0.f;

  const int nsteps = K >> 5;

#if USE_ASYNC
  auto stage = [&](int kk, int buf) {
    async_ld128(asrc + kk,     &Al[buf][arow * 32 + akh]);
    async_ld128(asrc + kk + 8, &Al[buf][arow * 32 + akh + 8]);
    async_ld128(bsrc + kk,     &Bl[buf][bcol * 32 + bkg]);
  };
  stage(0, 0);
  async_wait0();
  __syncthreads();
#else
  uint4 ra0, ra1, rb;
  auto gload = [&](int kk) {
    ra0 = *(const uint4*)(asrc + kk);
    ra1 = *(const uint4*)(asrc + kk + 8);
    rb  = *(const uint4*)(bsrc + kk);
    __builtin_prefetch(asrc + kk + 32, 0, 0);
  };
  auto sstore = [&](int buf) {
    *(uint4*)&Al[buf][arow * 32 + akh]     = ra0;
    *(uint4*)&Al[buf][arow * 32 + akh + 8] = ra1;
    *(uint4*)&Bl[buf][bcol * 32 + bkg]     = rb;
  };
  gload(0);
  sstore(0);
  __syncthreads();
#endif

  for (int step = 0; step < nsteps; ++step) {
    const int buf = step & 1;
#if USE_ASYNC
    if (step + 1 < nsteps) stage((step + 1) << 5, buf ^ 1);
#else
    if (step + 1 < nsteps) gload((step + 1) << 5);
#endif

    v16bf af[2], bfr[2];
    for (int sm = 0; sm < 2; ++sm) {
      int row = wm * 32 + sm * 16 + (lane & 15);
      const __bf16* p = &Al[buf][row * 32];
      af[sm] = pack_bf16(*(const uint4*)(p + hi), *(const uint4*)(p + 16 + hi));
    }
    for (int sn = 0; sn < 2; ++sn) {
      int col = wn * 32 + sn * 16 + (lane & 15);
      const __bf16* p = &Bl[buf][col * 32];
      bfr[sn] = pack_bf16(*(const uint4*)(p + hi), *(const uint4*)(p + 16 + hi));
    }
    for (int sm = 0; sm < 2; ++sm)
      for (int sn = 0; sn < 2; ++sn)
        acc[sm][sn] = wmma_bf16(af[sm], bfr[sn], acc[sm][sn]);

#if USE_ASYNC
    if (step + 1 < nsteps) async_wait0();
#else
    if (step + 1 < nsteps) sstore(buf ^ 1);
#endif
    __syncthreads();
  }

  // epilogue (C layout: VGPR r -> row r + hi, lane&15 -> column)
  for (int sm = 0; sm < 2; ++sm)
    for (int sn = 0; sn < 2; ++sn) {
      int colg = n0 + wn * 32 + sn * 16 + (lane & 15);
      for (int r = 0; r < 8; ++r) {
        int rowg = m0 + wm * 32 + sm * 16 + r + hi;
        float v = acc[sm][sn][r];
        if (BIAS) v += bias[colg];
        if (RES)  v += res[(size_t)rowg * N + colg];
        if (RELU) v = fmaxf(v, 0.f);
        if (OF32) outf[(size_t)rowg * N + colg] = v;
        if (OBF)  outb[(size_t)rowg * N + colg] = (__bf16)v;
        if (OBT) {  // [B,H,DHEAD,S] d-major layout for V
          int bb_ = rowg >> 11, s_ = rowg & (S_LEN - 1);
          int h_  = colg >> 7,  d_ = colg & (DHEAD - 1);
          outb[(((size_t)bb_ * NHEAD + h_) * DHEAD + d_) * S_LEN + s_] = (__bf16)v;
        }
      }
    }
}

// ---------------------------------------------------------------------------
// Flash attention with sim addend, causal + key-pad mask.
// Grid (S/64, H, B), 128 threads = 4 waves, wave = 16 query rows.
// K, V^T (pre-transposed by the V GEMM) and sim tiles all staged with async
// b128 copies; P re-striped through wave-private LDS between WMMA stages.
// ---------------------------------------------------------------------------
__global__ __launch_bounds__(128)
void attn_kernel(const __bf16* __restrict__ qb, const __bf16* __restrict__ kb,
                 const __bf16* __restrict__ vT, const int* __restrict__ et,
                 const float* __restrict__ simg, __bf16* __restrict__ ob) {
  __shared__ __align__(16) __bf16 Kl[64 * 128];    // [key][d]
  __shared__ __align__(16) __bf16 Vtl[128 * 64];   // [d][key]
  __shared__ __align__(16) float  Siml[64 * 64];   // [q_local][key_local]
  __shared__ float padl[64];
  __shared__ __align__(16) __bf16 Pl[4][16 * 32];  // per-wave P relayout buffer

  const int tid  = threadIdx.x;
  const int wave = tid >> 5;
  const int lane = tid & 31;
  const int hi   = (lane >> 4) * 8;          // 0 or 8
  const int b  = blockIdx.z;
  const int hh = blockIdx.y;
  const int q0 = blockIdx.x * 64;
  const int HD = NHEAD * DHEAD;              // 512

  // Q fragments for this wave's 16 rows (A-operand layout), 4 chunks of K=32
  v16bf qf[4];
  {
    int qrA = q0 + wave * 16 + (lane & 15);
    const __bf16* qp = qb + (size_t)(b * S_LEN + qrA) * HD + hh * DHEAD;
    for (int c = 0; c < 4; ++c)
      qf[c] = pack_bf16(*(const uint4*)(qp + c * 32 + hi),
                        *(const uint4*)(qp + c * 32 + 16 + hi));
  }

  v8f oacc[8];
  float mrun[8], lrun[8];
  for (int dt = 0; dt < 8; ++dt)
    for (int r = 0; r < 8; ++r) oacc[dt][r] = 0.f;
  for (int r = 0; r < 8; ++r) { mrun[r] = -INFINITY; lrun[r] = 0.f; }

  const int ntiles = blockIdx.x + 1;    // causal: keys only up to q0+63
  for (int t = 0; t < ntiles; ++t) {
    const int k0 = t * 64;
    __syncthreads();   // previous tile fully consumed before restaging
    {   // stage K tile rows as-is: Kl[key][d]
      int key = tid >> 1, half = (tid & 1) * 64;
      const __bf16* src = kb + (size_t)(b * S_LEN + k0 + key) * HD +
                          hh * DHEAD + half;
      __bf16* dst = &Kl[key * 128 + half];
#if USE_ASYNC
      for (int j = 0; j < 8; ++j) async_ld128(src + j * 8, dst + j * 8);
#else
      for (int j = 0; j < 8; ++j)
        ((uint4*)dst)[j] = ((const uint4*)src)[j];
#endif
    }
    {   // stage V^T tile: Vtl[d][key] straight from the d-major V buffer
      int d = tid;   // 0..127
      const __bf16* src = vT + (((size_t)b * NHEAD + hh) * DHEAD + d) * S_LEN + k0;
      __bf16* dst = &Vtl[d * 64];
#if USE_ASYNC
      for (int j = 0; j < 4; ++j) async_ld128(src + j * 8, dst + j * 8);
#else
      for (int j = 0; j < 4; ++j)
        ((uint4*)dst)[j] = ((const uint4*)src)[j];
#endif
    }
    {   // stage sim tile rows: Siml[q_local][key_local]
      int qi = tid >> 1, half = (tid & 1) * 32;
      const float* src = simg + ((size_t)(b * S_LEN + q0 + qi)) * S_LEN +
                         k0 + half;
      float* dst = &Siml[qi * 64 + half];
#if USE_ASYNC
      for (int j = 0; j < 8; ++j) async_ld128(src + j * 4, dst + j * 4);
#else
      for (int j = 0; j < 8; ++j)
        ((uint4*)dst)[j] = ((const uint4*)src)[j];
#endif
    }
    if (tid < 64) padl[tid] = (et[b * S_LEN + k0 + tid] == 0) ? -1e9f : 0.f;
#if USE_ASYNC
    async_wait0();
#endif
    __syncthreads();

    for (int kc = 0; kc < 64; kc += 32) {
      // ---- scores for 16 queries x 32 keys (two 16x16 tiles) ----
      v8f sc[2];
      for (int j16 = 0; j16 < 2; ++j16) {
        int kloc = kc + j16 * 16 + (lane & 15);
        // preload all four K fragments + the 8 sim values + pad addend so the
        // LDS loads pipeline underneath the WMMA chain
        v16bf kf[4];
        const __bf16* kp = &Kl[kloc * 128];
        for (int c = 0; c < 4; ++c) {
          const __bf16* p = kp + c * 32;
          kf[c] = pack_bf16(*(const uint4*)(p + hi), *(const uint4*)(p + 16 + hi));
        }
        float simv[8];
        for (int r = 0; r < 8; ++r)
          simv[r] = Siml[(wave * 16 + r + hi) * 64 + kloc];
        float padv = padl[kloc];

        for (int r = 0; r < 8; ++r) sc[j16][r] = 0.f;
        for (int c = 0; c < 4; ++c) sc[j16] = wmma_bf16(qf[c], kf[c], sc[j16]);

        for (int r = 0; r < 8; ++r) {
          int gi = q0 + wave * 16 + r + hi;
          int gj = k0 + kloc;
          float v = sc[j16][r] * 0.08838834764831845f + simv[r] + padv;
          sc[j16][r] = (gj > gi) ? -1e9f : v;
        }
      }
      // ---- online softmax update over this 32-key chunk ----
      float scale[8];
      for (int r = 0; r < 8; ++r) {
        float rm = fmaxf(sc[0][r], sc[1][r]);
        rm = fmaxf(rm, __shfl_xor(rm, 1, 32));
        rm = fmaxf(rm, __shfl_xor(rm, 2, 32));
        rm = fmaxf(rm, __shfl_xor(rm, 4, 32));
        rm = fmaxf(rm, __shfl_xor(rm, 8, 32));
        float mnew = fmaxf(mrun[r], rm);
        scale[r] = __expf(mrun[r] - mnew);
        float p0 = __expf(sc[0][r] - mnew);
        float p1 = __expf(sc[1][r] - mnew);
        sc[0][r] = p0; sc[1][r] = p1;
        float ps = p0 + p1;
        ps += __shfl_xor(ps, 1, 32);
        ps += __shfl_xor(ps, 2, 32);
        ps += __shfl_xor(ps, 4, 32);
        ps += __shfl_xor(ps, 8, 32);
        lrun[r] = lrun[r] * scale[r] + ps;
        mrun[r] = mnew;
      }
      for (int dt = 0; dt < 8; ++dt)
        for (int r = 0; r < 8; ++r) oacc[dt][r] *= scale[r];

      // ---- re-stripe P (C layout -> A layout) through wave-private LDS ----
      for (int j16 = 0; j16 < 2; ++j16)
        for (int r = 0; r < 8; ++r)
          Pl[wave][(r + hi) * 32 + j16 * 16 + (lane & 15)] = (__bf16)sc[j16][r];
      const __bf16* pp = &Pl[wave][(lane & 15) * 32];
      v16bf pf = pack_bf16(*(const uint4*)(pp + hi), *(const uint4*)(pp + 16 + hi));

      // ---- O += P @ V  (8 d-tiles of 16, V fragments preloaded in groups) ----
      for (int dt = 0; dt < 8; dt += 4) {
        v16bf vf[4];
        for (int g = 0; g < 4; ++g) {
          int d = (dt + g) * 16 + (lane & 15);
          const __bf16* p = &Vtl[d * 64 + kc];
          vf[g] = pack_bf16(*(const uint4*)(p + hi), *(const uint4*)(p + 16 + hi));
        }
        for (int g = 0; g < 4; ++g)
          oacc[dt + g] = wmma_bf16(pf, vf[g], oacc[dt + g]);
      }
    }
  }

  // ---- finalize: O /= l, store bf16 ----
  float inv[8];
  for (int r = 0; r < 8; ++r) inv[r] = (lrun[r] > 0.f) ? 1.f / lrun[r] : 0.f;
  for (int dt = 0; dt < 8; ++dt)
    for (int r = 0; r < 8; ++r) {
      int gi = q0 + wave * 16 + r + hi;
      ob[(size_t)(b * S_LEN + gi) * HD + hh * DHEAD + dt * 16 + (lane & 15)] =
          (__bf16)(oacc[dt][r] * inv[r]);
    }
}

// ---------------------------------------------------------------------------
// LayerNorm over 512 columns, one wave per row; y = ln(x)*g+b, * non_pad mask
// float4-vectorized loads/stores.
// ---------------------------------------------------------------------------
template<bool WF32, bool WBF>
__global__ __launch_bounds__(256)
void ln_kernel(const float* __restrict__ x, const float* __restrict__ g,
               const float* __restrict__ bb, const float* __restrict__ msk,
               float* __restrict__ of, __bf16* __restrict__ obf) {
  int wave = threadIdx.x >> 5, lane = threadIdx.x & 31;
  int row = blockIdx.x * 8 + wave;
  const float4* xr4 = (const float4*)(x + (size_t)row * DMODEL);
  float4 v4[4];
  float s = 0.f, sq = 0.f;
  for (int i = 0; i < 4; ++i) {
    v4[i] = xr4[lane + i * 32];
    const float* c = &v4[i].x;
    for (int q = 0; q < 4; ++q) { s += c[q]; sq += c[q] * c[q]; }
  }
  for (int m = 1; m < 32; m <<= 1) {
    s  += __shfl_xor(s,  m, 32);
    sq += __shfl_xor(sq, m, 32);
  }
  float mean = s * (1.f / DMODEL);
  float var  = sq * (1.f / DMODEL) - mean * mean;
  float rstd = rsqrtf(var + 1e-5f);
  float mk = msk[row];
  for (int i = 0; i < 4; ++i) {
    int c4 = lane + i * 32;
    float4 g4 = ((const float4*)g)[c4];
    float4 b4 = ((const float4*)bb)[c4];
    const float* vv = &v4[i].x;
    const float* gg = &g4.x;
    const float* bbq = &b4.x;
    float4 y4;
    float* yy = &y4.x;
    union { __bf16 h[4]; uint2 u; } pk;
    for (int q = 0; q < 4; ++q) {
      yy[q] = ((vv[q] - mean) * rstd * gg[q] + bbq[q]) * mk;
      pk.h[q] = (__bf16)yy[q];
    }
    if (WF32) ((float4*)(of + (size_t)row * DMODEL))[c4] = y4;
    if (WBF)  ((uint2*)(obf + (size_t)row * DMODEL))[c4] = pk.u;
  }
}

// ---------------------------------------------------------------------------
// host-side orchestration
// ---------------------------------------------------------------------------
extern "C" void kernel_launch(void* const* d_in, const int* in_sizes, int n_in,
                              void* d_out, int out_size, void* d_ws, size_t ws_size,
                              hipStream_t stream) {
  const int*   event_type = (const int*)  d_in[0];
  const int*   vertex     = (const int*)  d_in[1];
  const float* event_time = (const float*)d_in[2];
  const float* non_pad    = (const float*)d_in[3];
  const float* Amat       = (const float*)d_in[4];
  const float* Wmat       = (const float*)d_in[5];
  const float* eemb       = (const float*)d_in[6];
  const float* vemb       = (const float*)d_in[7];
  const float* Wq   = (const float*)d_in[8];
  const float* Wk   = (const float*)d_in[9];
  const float* Wv   = (const float*)d_in[10];
  const float* Wo   = (const float*)d_in[11];
  const float* bo   = (const float*)d_in[12];
  const float* ln1g = (const float*)d_in[13];
  const float* ln1b = (const float*)d_in[14];
  const float* w1   = (const float*)d_in[15];
  const float* b1   = (const float*)d_in[16];
  const float* w2   = (const float*)d_in[17];
  const float* b2   = (const float*)d_in[18];
  const float* ln2g = (const float*)d_in[19];
  const float* ln2b = (const float*)d_in[20];

  float* enc_out = (float*)d_out;                       // [8192, 512]
  float* sim_out = enc_out + (size_t)ROWS * DMODEL;     // [4, 2048, 2048]

  char* ws = (char*)d_ws;
  size_t off = 0;
  auto carve = [&](size_t bytes) -> void* {
    off = (off + 255) & ~(size_t)255;
    void* p = ws + off;
    off += bytes;
    return p;
  };

  float*  enc0f = (float*) carve((size_t)ROWS * 512 * 4);
  __bf16* enc0b = (__bf16*)carve((size_t)ROWS * 512 * 2);
  __bf16* wqb   = (__bf16*)carve((size_t)512 * 512 * 2);    // [N][K]
  __bf16* wkb   = (__bf16*)carve((size_t)512 * 512 * 2);
  __bf16* wvb   = (__bf16*)carve((size_t)512 * 512 * 2);
  __bf16* wob   = (__bf16*)carve((size_t)512 * 512 * 2);
  __bf16* w1t   = (__bf16*)carve((size_t)1024 * 512 * 2);   // [1024][512]
  __bf16* w2t   = (__bf16*)carve((size_t)512 * 1024 * 2);   // [512][1024]
  __bf16* qbuf  = (__bf16*)carve((size_t)ROWS * 512 * 2);
  __bf16* kbuf  = (__bf16*)carve((size_t)ROWS * 512 * 2);
  __bf16* vtbuf = (__bf16*)carve((size_t)ROWS * 512 * 2);   // [B,H,DHEAD,S]
  __bf16* obuf  = (__bf16*)carve((size_t)ROWS * 512 * 2);
  float*  xbuf  = (float*) carve((size_t)ROWS * 512 * 4);   // reused x1 & x2
  float*  enc1f = (float*) carve((size_t)ROWS * 512 * 4);
  __bf16* enc1b = (__bf16*)carve((size_t)ROWS * 512 * 2);
  __bf16* hbuf  = (__bf16*)carve((size_t)ROWS * 1024 * 2);
  (void)ws_size; (void)n_in; (void)in_sizes; (void)out_size;

  // weights -> bf16, transposed to [N][K]
  cvtT_kernel<<<dim3(16, 16), 256, 0, stream>>>(Wq, wqb, 512, 512);
  cvtT_kernel<<<dim3(16, 16), 256, 0, stream>>>(Wk, wkb, 512, 512);
  cvtT_kernel<<<dim3(16, 16), 256, 0, stream>>>(Wv, wvb, 512, 512);
  cvtT_kernel<<<dim3(16, 16), 256, 0, stream>>>(Wo, wob, 512, 512);
  cvtT_kernel<<<dim3(16, 32), 256, 0, stream>>>(w1, w1t, 512, 1024);
  cvtT_kernel<<<dim3(32, 16), 256, 0, stream>>>(w2, w2t, 1024, 512);

  // embeddings + temporal encoding
  embed_kernel<<<ROWS, 256, 0, stream>>>(event_type, vertex, event_time, non_pad,
                                         eemb, vemb, enc0f, enc0b);

  // sim -> output slice (also consumed by attention)
  sim_kernel<<<(int)(((size_t)4 * S_LEN * S_LEN) / 1024), 256, 0, stream>>>(
      vertex, Amat, Wmat, sim_out);

  // Q/K/V projections (V written d-major for transpose-free attention staging)
  dim3 g512(512 / 64, ROWS / 128);
  gemm_bf16_kernel<false, false, false, false, true, false><<<g512, 256, 0, stream>>>(
      enc0b, wqb, nullptr, nullptr, nullptr, qbuf, ROWS, 512, 512);
  gemm_bf16_kernel<false, false, false, false, true, false><<<g512, 256, 0, stream>>>(
      enc0b, wkb, nullptr, nullptr, nullptr, kbuf, ROWS, 512, 512);
  gemm_bf16_kernel<false, false, false, false, false, true><<<g512, 256, 0, stream>>>(
      enc0b, wvb, nullptr, nullptr, nullptr, vtbuf, ROWS, 512, 512);

  // attention
  attn_kernel<<<dim3(S_LEN / 64, NHEAD, 4), 128, 0, stream>>>(
      qbuf, kbuf, vtbuf, event_type, sim_out, obuf);

  // output projection + bias + residual -> x1 (f32)
  gemm_bf16_kernel<true, false, true, true, false, false><<<g512, 256, 0, stream>>>(
      obuf, wob, bo, enc0f, xbuf, nullptr, ROWS, 512, 512);

  // LN1 (+ pad mask) -> enc1 f32 + bf16
  ln_kernel<true, true><<<ROWS / 8, 256, 0, stream>>>(xbuf, ln1g, ln1b, non_pad,
                                                      enc1f, enc1b);

  // FFN: relu(enc1 @ w1 + b1) @ w2 + b2 + enc1
  dim3 g1024(1024 / 64, ROWS / 128);
  gemm_bf16_kernel<true, true, false, false, true, false><<<g1024, 256, 0, stream>>>(
      enc1b, w1t, b1, nullptr, nullptr, hbuf, ROWS, 1024, 512);
  gemm_bf16_kernel<true, false, true, true, false, false><<<g512, 256, 0, stream>>>(
      hbuf, w2t, b2, enc1f, xbuf, nullptr, ROWS, 512, 1024);

  // LN2 (+ pad mask) -> final enc output (f32)
  ln_kernel<true, false><<<ROWS / 8, 256, 0, stream>>>(xbuf, ln2g, ln2b, non_pad,
                                                       enc_out, nullptr);
}
